// CausalAttention_23115513987673
// MI455X (gfx1250) — matrix operand order repaired and checked by reading, not compiled, not run on hardware
//
#include <hip/hip_runtime.h>
#include <math.h>

// Causal single-head attention, B=4 S=2048 D=768, fp32 in/out.
// f16 WMMA (v_wmma_f32_16x16x32_f16) with f32 accumulation.
// BK=64 (8 WMMA/wave per sync), A tile staged via gfx1250 async global->LDS
// copies (ASYNCcnt); fixed pointer types for the async builtin (int4 vectors).

constexpr int BATCH = 4;
constexpr int SEQ   = 2048;
constexpr int DIM   = 768;

typedef __attribute__((ext_vector_type(16))) _Float16 v16h;
typedef __attribute__((ext_vector_type(8)))  float    v8f;

struct alignas(16) h8 { _Float16 h[8]; };

#if __has_builtin(__builtin_amdgcn_global_load_async_to_lds_b128)
#define USE_ASYNC_LDS 1
#else
#define USE_ASYNC_LDS 0
#endif

#if USE_ASYNC_LDS
// Builtin signature (from hipcc diagnostic): first param is a pointer to
// 'int __attribute__((__vector_size__(16)))' in AS(1); LDS side in AS(3).
typedef int v4i_vs __attribute__((vector_size(16)));
typedef __attribute__((address_space(1))) v4i_vs g_v4i;
typedef __attribute__((address_space(3))) v4i_vs l_v4i;

// 16B per-lane async copy, tracked by ASYNCcnt (not DScnt!).
static __device__ __forceinline__ void async_cp16(_Float16* lds, const _Float16* g) {
  __builtin_amdgcn_global_load_async_to_lds_b128(
      (g_v4i*)(void*)g, (l_v4i*)(void*)lds, /*offset=*/0, /*cpol=*/0);
}
static __device__ __forceinline__ void wait_async0() {
#if __has_builtin(__builtin_amdgcn_s_wait_asynccnt)
  __builtin_amdgcn_s_wait_asynccnt(0);
#else
  asm volatile("s_wait_asynccnt 0x0" ::: "memory");
#endif
}
#endif

static __device__ __forceinline__ v8f wmma16(v16h a, v16h b, v8f c) {
  return __builtin_amdgcn_wmma_f32_16x16x32_f16(
      /*neg_a=*/false, a, /*neg_b=*/false, b,
      /*c_mod=*/(short)0, c, /*reuse_a=*/false, /*reuse_b=*/false);
}

// A fragment (16x32, row-major source, leading dim lda in halves).
// lane L holds row M=L&15; K base 0 (lanes 0-15) / 16 (lanes 16-31);
// elems 0..15 -> K=kb..kb+15 (contiguous).  Two 16B loads.
static __device__ __forceinline__ v16h load_afrag(const _Float16* p, int lda) {
  int lane = threadIdx.x & 31;
  int row  = lane & 15;
  int kb   = (lane & 16) ? 16 : 0;
  const _Float16* q = p + (size_t)row * lda + kb;
  union { h8 s[2]; v16h v; } u;
  u.s[0] = *(const h8*)(q);
  u.s[1] = *(const h8*)(q + 8);
  return u.v;
}

// B fragment for S = Q*K^T: b[e] = K[n][kb2+e] -> contiguous 32B per lane
// straight from row-major K.
static __device__ __forceinline__ v16h load_bfrag_kt(const _Float16* p, int ld) {
  int lane = threadIdx.x & 31;
  int n    = lane & 15;
  int kb2  = (lane & 16) ? 16 : 0;
  const _Float16* q = p + (size_t)n * ld + kb2;
  union { h8 s[2]; v16h v; } u;
  u.s[0] = *(const h8*)(q);
  u.s[1] = *(const h8*)(q + 8);
  return u.v;
}

// B fragment from a TRANSPOSED tile in LDS: BsT[n][k], ld = ldt, k-offset koff.
// Lane needs B[koff+kb2+e][col+n] = BsT[col+n][koff+kb2+e] -> contiguous 32B.
static __device__ __forceinline__ v16h load_bfrag_t(const _Float16* BsT, int ldt,
                                                    int col, int koff) {
  int lane = threadIdx.x & 31;
  int n    = col + (lane & 15);
  int kb2  = (lane & 16) ? 16 : 0;
  const _Float16* q = BsT + (size_t)n * ldt + koff + kb2;
  union { h8 s[2]; v16h v; } u;
  u.s[0] = *(const h8*)(q);
  u.s[1] = *(const h8*)(q + 8);
  return u.v;
}

// ---------------------------------------------------------------- convert ---
__global__ void f2h_kernel(const float* __restrict__ in, _Float16* __restrict__ out, int n) {
  int i = blockIdx.x * blockDim.x + threadIdx.x;
  if (i < n) out[i] = (_Float16)in[i];
}

// Stage a 64x64 f16 A tile (rows m0.., cols kk..kk+63) row-major into As.
// 256 threads x 16 halves; async global->LDS when available.
template <int LDA>
static __device__ __forceinline__ void stage_a64(_Float16* As, const _Float16* src,
                                                 int sld, int m0, int kk, int t) {
  int r = t >> 2;            // 0..63
  int c = (t & 3) * 16;      // 0,16,32,48
  const _Float16* g = &src[(size_t)(m0 + r) * sld + kk + c];
  _Float16* l = &As[r * LDA + c];
#if USE_ASYNC_LDS
  async_cp16(l, g);
  async_cp16(l + 8, g + 8);
#else
  *(h8*)l       = *(const h8*)g;
  *(h8*)(l + 8) = *(const h8*)(g + 8);
#endif
}

// Stage a 32x128 f16 tile (rows kk..kk+31, cols n0.. of src) TRANSPOSED into
// BsT[128][LDT] at k-offset koff.  Each thread: 2 consecutive rows x 8 cols,
// packs row pairs -> 8x ds_store_b32 (compiler merges to 2-addr stores).
template <int LDT>
static __device__ __forceinline__ void stage_bT(_Float16* BsT,
                                                const _Float16* src, int sld,
                                                int kk, int n0, int koff, int t) {
  int r = (t >> 4) * 2;        // 0,2,...,30
  int c = (t & 15) * 8;        // 0,8,...,120
  h8 lo = *(const h8*)&src[(size_t)(kk + r) * sld + n0 + c];
  h8 hi = *(const h8*)&src[(size_t)(kk + r + 1) * sld + n0 + c];
#pragma unroll
  for (int i = 0; i < 8; ++i) {
    union { _Float16 h[2]; unsigned u; } pk_;
    pk_.h[0] = lo.h[i];
    pk_.h[1] = hi.h[i];
    *(unsigned*)&BsT[(size_t)(c + i) * LDT + koff + r] = pk_.u;
  }
}

// ------------------------------------------------------------- QKV GEMM -----
// C[8192x768] = A[8192x768] * W[768x768], f16 in -> f16 out.
// Block: 256 thr (8 waves, 2x4), tile 64x128, BK=64 (2 sub-steps per stage).
__global__ void __launch_bounds__(256)
qkv_gemm_kernel(const _Float16* __restrict__ A,
                const _Float16* __restrict__ Wq,
                const _Float16* __restrict__ Wk,
                const _Float16* __restrict__ Wv,
                _Float16* __restrict__ Qh, _Float16* __restrict__ Kh,
                _Float16* __restrict__ Vh) {
  constexpr int LDA = 72;   // 64 + pad (rows of A tile)
  constexpr int LDT = 72;   // 64 + pad (transposed B: 128 rows x 64 k)
  __shared__ alignas(16) _Float16 As[64 * LDA];
  __shared__ alignas(16) _Float16 BsT[128 * LDT];

  const int m0 = blockIdx.x * 64;
  const int n0 = blockIdx.y * 128;
  const int z  = blockIdx.z;
  const _Float16* Wp = (z == 0) ? Wq : (z == 1) ? Wk : Wv;

  const int t    = threadIdx.x;
  const int w    = t >> 5;
  const int wm   = w >> 2;      // 0..1
  const int wn   = w & 3;       // 0..3
  const int lane = t & 31;

  v8f acc[2][2];
#pragma unroll
  for (int i = 0; i < 2; ++i)
#pragma unroll
    for (int j = 0; j < 2; ++j) acc[i][j] = (v8f){0,0,0,0,0,0,0,0};

  for (int kk = 0; kk < DIM; kk += 64) {
    stage_a64<LDA>(As, A, DIM, m0, kk, t);
    stage_bT<LDT>(BsT, Wp, DIM, kk,      n0,  0, t);
    stage_bT<LDT>(BsT, Wp, DIM, kk + 32, n0, 32, t);
#if USE_ASYNC_LDS
    wait_async0();   // __syncthreads only fences DScnt, not ASYNCcnt
#endif
    __syncthreads();

#pragma unroll
    for (int s = 0; s < 2; ++s) {
      v16h af[2];
#pragma unroll
      for (int i = 0; i < 2; ++i)
        af[i] = load_afrag(&As[(wm * 32 + i * 16) * LDA + s * 32], LDA);
#pragma unroll
      for (int j = 0; j < 2; ++j) {
        v16h bf = load_bfrag_t(BsT, LDT, wn * 32 + j * 16, s * 32);
#pragma unroll
        for (int i = 0; i < 2; ++i) acc[i][j] = wmma16(af[i], bf, acc[i][j]);
      }
    }
    __syncthreads();
  }

  _Float16* outp = (z == 0) ? Qh : (z == 1) ? Kh : Vh;
  const int rowg = lane >> 4, n = lane & 15;
#pragma unroll
  for (int i = 0; i < 2; ++i)
#pragma unroll
    for (int j = 0; j < 2; ++j)
#pragma unroll
      for (int v = 0; v < 8; ++v) {
        int r = m0 + wm * 32 + i * 16 + v + 8 * rowg;
        int c = n0 + wn * 32 + j * 16 + n;
        outp[(size_t)r * DIM + c] = (_Float16)acc[i][j][v];
      }
}

// --------------------------------------------------- scores + softmax -------
// grid (SEQ/64, BATCH), block 128 (4 independent waves; wave owns 16 q rows).
// Q fragments (24 x v16h = 192 VGPRs) are register-resident for the whole
// kernel (wave32 budget is up to 1024 VGPRs).
// Pass 1: row max of scaled masked scores. Pass 2: recompute, write
// P = exp(s - m) (f16, unnormalized) and linv = 1/rowsum.
__global__ void __launch_bounds__(128)
attn_scores_kernel(const _Float16* __restrict__ Qh,
                   const _Float16* __restrict__ Kh,
                   _Float16* __restrict__ P, float* __restrict__ linv) {
  constexpr int ND = DIM / 32;  // 24 d-steps
  const int qt = blockIdx.x, b = blockIdx.y;
  const int w = threadIdx.x >> 5, lane = threadIdx.x & 31;
  const int q0 = qt * 64 + w * 16;
  const _Float16* Qrow = Qh + ((size_t)b * SEQ + q0) * DIM;
  const _Float16* Kb   = Kh + (size_t)b * SEQ * DIM;
  _Float16* Pb = P + (size_t)b * SEQ * SEQ;
  const float scale = 1.0f / sqrtf((float)DIM);
  const int rowg = lane >> 4, n = lane & 15;

  v16h qf[ND];
#pragma unroll
  for (int d = 0; d < ND; ++d) qf[d] = load_afrag(Qrow + d * 32, DIM);

  float mreg[8];
#pragma unroll
  for (int v = 0; v < 8; ++v) mreg[v] = -3.0e38f;

  // ---- pass 1: max ----
  for (int kt = 0; kt <= qt; ++kt) {
    const int k0 = kt * 64;
    if (kt < qt)  // prefetch next K tile (global_prefetch_b8)
      __builtin_prefetch(Kb + (size_t)(k0 + 64) * DIM, 0, 0);
    v8f acc[4];
#pragma unroll
    for (int j = 0; j < 4; ++j) acc[j] = (v8f){0,0,0,0,0,0,0,0};
#pragma unroll
    for (int d = 0; d < ND; ++d) {
#pragma unroll
      for (int j = 0; j < 4; ++j) {
        v16h bb = load_bfrag_kt(Kb + (size_t)(k0 + j * 16) * DIM + d * 32, DIM);
        acc[j] = wmma16(qf[d], bb, acc[j]);
      }
    }
#pragma unroll
    for (int j = 0; j < 4; ++j)
#pragma unroll
      for (int v = 0; v < 8; ++v) {
        float s = acc[j][v] * scale;
        int qi = q0 + v + 8 * rowg;
        int ki = k0 + j * 16 + n;
        if (ki > qi) s = -3.0e38f;          // only bites on diagonal tile
        mreg[v] = fmaxf(mreg[v], s);
      }
  }
#pragma unroll
  for (int v = 0; v < 8; ++v)
#pragma unroll
    for (int mask = 1; mask < 16; mask <<= 1)
      mreg[v] = fmaxf(mreg[v], __shfl_xor(mreg[v], mask, 32));

  float sreg[8];
#pragma unroll
  for (int v = 0; v < 8; ++v) sreg[v] = 0.0f;

  // ---- pass 2: exp + store P ----
  for (int kt = 0; kt <= qt; ++kt) {
    const int k0 = kt * 64;
    v8f acc[4];
#pragma unroll
    for (int j = 0; j < 4; ++j) acc[j] = (v8f){0,0,0,0,0,0,0,0};
#pragma unroll
    for (int d = 0; d < ND; ++d) {
#pragma unroll
      for (int j = 0; j < 4; ++j) {
        v16h bb = load_bfrag_kt(Kb + (size_t)(k0 + j * 16) * DIM + d * 32, DIM);
        acc[j] = wmma16(qf[d], bb, acc[j]);
      }
    }
#pragma unroll
    for (int j = 0; j < 4; ++j)
#pragma unroll
      for (int v = 0; v < 8; ++v) {
        int qi = q0 + v + 8 * rowg;
        int ki = k0 + j * 16 + n;
        float s = acc[j][v] * scale;
        float p = (ki > qi) ? 0.0f : __expf(s - mreg[v]);
        sreg[v] += p;
        Pb[(size_t)qi * SEQ + ki] = (_Float16)p;
      }
  }
#pragma unroll
  for (int v = 0; v < 8; ++v)
#pragma unroll
    for (int mask = 1; mask < 16; mask <<= 1)
      sreg[v] += __shfl_xor(sreg[v], mask, 32);
  if (n == 0) {
#pragma unroll
    for (int v = 0; v < 8; ++v)
      linv[b * SEQ + q0 + v + 8 * rowg] = 1.0f / sreg[v];
  }
}

// ------------------------------------------------------------- P*V GEMM -----
// O[b] = (P[b] * V[b]) scaled by linv; k-loop bounded by causal diagonal.
__global__ void __launch_bounds__(256)
pv_gemm_kernel(const _Float16* __restrict__ P,
               const _Float16* __restrict__ Vh,
               const float* __restrict__ linv,
               float* __restrict__ out) {
  constexpr int LDA = 72;
  constexpr int LDT = 72;
  __shared__ alignas(16) _Float16 As[64 * LDA];
  __shared__ alignas(16) _Float16 BsT[128 * LDT];

  const int m0 = blockIdx.x * 64;            // q rows within batch
  const int n0 = blockIdx.y * 128;           // dim cols
  const int b  = blockIdx.z;
  const _Float16* Pb = P  + (size_t)b * SEQ * SEQ;
  const _Float16* Vb = Vh + (size_t)b * SEQ * DIM;
  float* Ob = out + (size_t)b * SEQ * DIM;
  const int klim = m0 + 64;                  // causal: keys <= last q row

  const int t    = threadIdx.x;
  const int w    = t >> 5;
  const int wm   = w >> 2;
  const int wn   = w & 3;
  const int lane = t & 31;

  v8f acc[2][2];
#pragma unroll
  for (int i = 0; i < 2; ++i)
#pragma unroll
    for (int j = 0; j < 2; ++j) acc[i][j] = (v8f){0,0,0,0,0,0,0,0};

  for (int kk = 0; kk < klim; kk += 64) {
    stage_a64<LDA>(As, Pb, SEQ, m0, kk, t);
    stage_bT<LDT>(BsT, Vb, DIM, kk,      n0,  0, t);
    stage_bT<LDT>(BsT, Vb, DIM, kk + 32, n0, 32, t);
#if USE_ASYNC_LDS
    wait_async0();
#endif
    __syncthreads();

#pragma unroll
    for (int s = 0; s < 2; ++s) {
      v16h af[2];
#pragma unroll
      for (int i = 0; i < 2; ++i)
        af[i] = load_afrag(&As[(wm * 32 + i * 16) * LDA + s * 32], LDA);
#pragma unroll
      for (int j = 0; j < 2; ++j) {
        v16h bf = load_bfrag_t(BsT, LDT, wn * 32 + j * 16, s * 32);
#pragma unroll
        for (int i = 0; i < 2; ++i) acc[i][j] = wmma16(af[i], bf, acc[i][j]);
      }
    }
    __syncthreads();
  }

  const int rowg = lane >> 4, n = lane & 15;
#pragma unroll
  for (int i = 0; i < 2; ++i)
#pragma unroll
    for (int j = 0; j < 2; ++j)
#pragma unroll
      for (int v = 0; v < 8; ++v) {
        int r = m0 + wm * 32 + i * 16 + v + 8 * rowg;
        int c = n0 + wn * 32 + j * 16 + n;
        Ob[(size_t)r * DIM + c] = acc[i][j][v] * linv[b * SEQ + r];
      }
}

// ----------------------------------------------------------------- launch ---
extern "C" void kernel_launch(void* const* d_in, const int* in_sizes, int n_in,
                              void* d_out, int out_size, void* d_ws, size_t ws_size,
                              hipStream_t stream) {
  const float* x  = (const float*)d_in[0];
  const float* Wq = (const float*)d_in[1];
  const float* Wk = (const float*)d_in[2];
  const float* Wv = (const float*)d_in[3];
  float* out = (float*)d_out;

  // Workspace bump allocator (~84 MB total).
  char* wp = (char*)d_ws;
  auto alloc = [&](size_t bytes) -> void* {
    void* p = (void*)wp;
    wp += (bytes + 255) & ~(size_t)255;
    return p;
  };
  _Float16* xh  = (_Float16*)alloc((size_t)BATCH * SEQ * DIM * sizeof(_Float16));
  _Float16* wqh = (_Float16*)alloc((size_t)DIM * DIM * sizeof(_Float16));
  _Float16* wkh = (_Float16*)alloc((size_t)DIM * DIM * sizeof(_Float16));
  _Float16* wvh = (_Float16*)alloc((size_t)DIM * DIM * sizeof(_Float16));
  _Float16* Qh  = (_Float16*)alloc((size_t)BATCH * SEQ * DIM * sizeof(_Float16));
  _Float16* Kh  = (_Float16*)alloc((size_t)BATCH * SEQ * DIM * sizeof(_Float16));
  _Float16* Vh  = (_Float16*)alloc((size_t)BATCH * SEQ * DIM * sizeof(_Float16));
  _Float16* P   = (_Float16*)alloc((size_t)BATCH * SEQ * SEQ * sizeof(_Float16));
  float*    li  = (float*)alloc((size_t)BATCH * SEQ * sizeof(float));

  // 1) fp32 -> f16 conversions
  {
    int n = BATCH * SEQ * DIM;
    f2h_kernel<<<(n + 255) / 256, 256, 0, stream>>>(x, xh, n);
    int m = DIM * DIM;
    f2h_kernel<<<(m + 255) / 256, 256, 0, stream>>>(Wq, wqh, m);
    f2h_kernel<<<(m + 255) / 256, 256, 0, stream>>>(Wk, wkh, m);
    f2h_kernel<<<(m + 255) / 256, 256, 0, stream>>>(Wv, wvh, m);
  }

  // 2) Q/K/V projections (WMMA GEMM), z selects which weight/output
  qkv_gemm_kernel<<<dim3(BATCH * SEQ / 64, DIM / 128, 3), 256, 0, stream>>>(
      xh, wqh, wkh, wvh, Qh, Kh, Vh);

  // 3) causal scores + softmax stats -> P (unnormalized), linv
  attn_scores_kernel<<<dim3(SEQ / 64, BATCH), 128, 0, stream>>>(Qh, Kh, P, li);

  // 4) O = (P @ V) * linv
  pv_gemm_kernel<<<dim3(SEQ / 64, DIM / 128, BATCH), 256, 0, stream>>>(
      P, Vh, li, out);
}
